// NoiseMoELayer_5918464934325
// MI455X (gfx1250) — matrix operand order repaired and checked by reading, not compiled, not run on hardware
//
#include <hip/hip_runtime.h>
#include <cstdint>

// ---------------------------------------------------------------------------
// NoiseMoE (B=32, C=128, H=W=56, E=4) for gfx1250 / MI455X.
// Heavy convs: implicit GEMM on v_wmma_f32_16x16x32_bf16 (wave32), 64-pos
// N-tiles (4 sub-tiles per A-fragment load). 1x1 expert GEMMs stage B-tiles
// with the Tensor Data Mover (double-buffered, TENSORcnt-synchronized).
// ---------------------------------------------------------------------------

typedef __attribute__((ext_vector_type(16))) __bf16   v16bf;
typedef __attribute__((ext_vector_type(8)))  float    v8f;
typedef __attribute__((ext_vector_type(4)))  unsigned v4u;
typedef __attribute__((ext_vector_type(8)))  unsigned v8u;

#define B_N   32
#define C_N   128
#define W_N   56
#define HW_N  3136
#define NT_N  49                       // HW / 64 spatial tiles
#define NPIX  (B_N * C_N * HW_N)       // 12,845,056

// ---- workspace layout (bytes, 256-aligned). Total ~181.4 MB ----
#define OFS_XBF   ((size_t)0)                              // x as bf16 (NCHW)
#define OFS_PACKS (OFS_XBF   + (size_t)NPIX * 2)           // shared 3x3 A-frags
#define OFS_PACKC (OFS_PACKS + 294912)                     // cd-effective 3x3
#define OFS_PACKB (OFS_PACKC + 294912)                     // bayar 5x5
#define OFS_PACKR (OFS_PACKB + 819200)                     // srm 1x1 (K=384)
#define OFS_PACKH (OFS_PACKR + 98304)                      // hf  1x1 (K=128)
#define OFS_SRM   (OFS_PACKH + 32768)                      // srm mid (B,HW,384) bf16
#define OFS_HFM   (OFS_SRM   + (size_t)B_N * 384 * HW_N * 2) // hf mid (B,HW,128) bf16
#define OFS_E2    (OFS_HFM   + (size_t)NPIX * 2)           // e2 raw bf16 (NCHW)
#define OFS_E3    (OFS_E2    + (size_t)NPIX * 2)           // e3 raw bf16 (NCHW)
#define OFS_BN    (OFS_E3    + (size_t)NPIX * 2)           // 512 floats scale/shift
#define OFS_EID   (OFS_BN    + 2048)                       // 32 ints

// ---------------------------------------------------------------------------
// TDM: 2D tile (tile_x contiguous elements per row, tile_y rows, row stride
// `stride` elements, 2-byte elements) global -> LDS. D# per ISA 08 §8.3/8.4.
__device__ __forceinline__ void tdm_load_2d(unsigned lds_off, const void* gptr,
                                            unsigned tile_x, unsigned tile_y,
                                            unsigned stride_elems) {
    uint64_t ga = (uint64_t)(uintptr_t)gptr;
    v4u g0 = { 1u,                                  // count=1, no gather
               lds_off,                             // lds_addr
               (unsigned)ga,                        // global_addr[31:0]
               (unsigned)((ga >> 32) & 0x1FFFFFFu) | (2u << 30) }; // addr hi | type=2
    v8u g1 = { 0x10000u,                            // data_size=1 (2B), mask=0
               (tile_x & 0xFFFFu) << 16,            // tensor_dim0 lo16
               (tile_y & 0xFFFFu) << 16,            // tdim0 hi | tensor_dim1 lo16
               (tile_x & 0xFFFFu) << 16,            // tdim1 hi | tile_dim0
               (tile_y & 0xFFFFu),                  // tile_dim1 | tile_dim2=0
               stride_elems,                        // tensor_dim0_stride lo32
               0u, 0u };
    asm volatile("tensor_load_to_lds %0, %1" :: "s"(g0), "s"(g1) : "memory");
}

// ---------------------------------------------------------------------------
__global__ void k_convert_bf16(const float* __restrict__ x, __bf16* __restrict__ xb) {
    for (int i = blockIdx.x * blockDim.x + threadIdx.x; i < NPIX;
         i += gridDim.x * blockDim.x)
        xb[i] = (__bf16)x[i];
}

// Pack weights into A-fragments in contiguous k order (matches the LDS B reads):
// lane l (element s = 0..15): M = l&15, K = s + (l>=16 ? 16 : 0)
__global__ void k_pack(const float* __restrict__ src, __bf16* __restrict__ dst,
                       int mode, int NK) {
    int total = 8 * NK * 512;
    for (int idx = blockIdx.x * blockDim.x + threadIdx.x; idx < total;
         idx += gridDim.x * blockDim.x) {
        int hh   = idx & 15;
        int lane = (idx >> 4) & 31;
        int kc   = (idx >> 9) % NK;
        int mt   = idx / (512 * NK);
        int M    = 16 * mt + (lane & 15);
        int Kl   = hh + ((lane & 16) ? 16 : 0);
        int K    = 32 * kc + Kl;
        float v = 0.f;
        if (mode == 0) {                    // shared 3x3, K = tap*128 + c
            int t = K >> 7, c = K & 127;
            v = src[(size_t)(M * 128 + c) * 9 + t];
        } else if (mode == 1) {             // cd-effective 3x3
            int t = K >> 7, c = K & 127;
            const float* wr = src + (size_t)(M * 128 + c) * 9;
            v = wr[t];
            if (t == 4) {
                float s = 0.f;
                #pragma unroll
                for (int j = 0; j < 9; ++j) s += wr[j];
                v -= 0.7f * s;
            }
        } else if (mode == 2) {             // bayar-constrained 5x5
            int t = K >> 7, c = K & 127;
            const float* rr = src + (size_t)(M * 128 + c) * 24;
            if (t == 12) v = -1.f;
            else {
                float s = 0.f;
                #pragma unroll
                for (int j = 0; j < 24; ++j) s += rr[j];
                v = rr[t < 12 ? t : t - 1] / s;
            }
        } else if (mode == 3) {             // srm 1x1, K=0..383
            v = src[(size_t)M * 384 + K];
        } else {                            // hf 1x1, K=0..127
            v = src[(size_t)M * 128 + K];
        }
        dst[idx] = (__bf16)v;
    }
}

// ---------------------------------------------------------------------------
// Gating: pooled mean -> logits -> top-1 (softmax degenerates to 1.0)
__global__ void k_gate(const float* __restrict__ x, const float* __restrict__ Wg,
                       float* __restrict__ out_w, int* __restrict__ eid) {
    int b = blockIdx.x;
    __shared__ float part[256];
    __shared__ float pooled[C_N];
    __shared__ float slog[4];
    int tid = threadIdx.x;
    int c = tid >> 1, half = tid & 1;
    const float* xp = x + (size_t)(b * C_N + c) * HW_N;
    float s = 0.f;
    for (int p = half * 1568; p < half * 1568 + 1568; ++p) s += xp[p];
    part[tid] = s;
    __syncthreads();
    if (half == 0) pooled[c] = (part[2 * c] + part[2 * c + 1]) * (1.0f / 3136.0f);
    if (tid < 4) slog[tid] = 0.f;
    __syncthreads();
    if (tid < C_N) {
        float pc = pooled[tid];
        #pragma unroll
        for (int e = 0; e < 4; ++e) atomicAdd(&slog[e], pc * Wg[tid * 4 + e]);
    }
    __syncthreads();
    if (tid == 0) {
        int best = 0; float bv = slog[0];
        for (int e = 1; e < 4; ++e) if (slog[e] > bv) { bv = slog[e]; best = e; }
        eid[b] = best;
        for (int e = 0; e < 4; ++e) out_w[b * 4 + e] = (e == best) ? 1.f : 0.f;
    }
}

__global__ void k_gate_fin(const float* __restrict__ w, float* __restrict__ tail) {
    if (blockIdx.x == 0 && threadIdx.x == 0) {
        float imp[4] = {0, 0, 0, 0};
        for (int b = 0; b < B_N; ++b)
            for (int e = 0; e < 4; ++e) imp[e] += w[b * 4 + e];
        float m = 0.25f * (imp[0] + imp[1] + imp[2] + imp[3]);
        float v = 0.f;
        for (int e = 0; e < 4; ++e) { float d = imp[e] - m; v += d * d; }
        v *= 0.25f;
        tail[0] = v / (m * m + 1e-10f);                    // aux_loss
        for (int e = 0; e < 4; ++e) tail[1 + e] = imp[e];  // examples_per_expert
        for (int e = 0; e < 4; ++e) tail[5 + e] = imp[e];  // expert_importance
    }
}

// ---------------------------------------------------------------------------
// Main WMMA conv: shared 3x3 always; fused cd (expert 0); bayar 5x5 (expert 1).
// WG = 8 waves; wave w owns out-channels [16w,16w+16); N-tile = 64 positions
// (4 sub-tiles share each A-fragment load).
__global__ void __launch_bounds__(256, 2)
k_conv_main(const __bf16* __restrict__ xb, const __bf16* __restrict__ packS,
            const __bf16* __restrict__ packC, const __bf16* __restrict__ packB,
            const int* __restrict__ eid, float* __restrict__ out) {
    __shared__ __align__(32) __bf16 tb[9][64][32];
    int wg = blockIdx.x;
    int b = wg / NT_N, nt = wg % NT_N;
    int p0 = nt * 64;
    int tid = threadIdx.x;
    int lane = tid & 31, wave = tid >> 5;
    int nOff = lane & 15;
    int cOff = lane & 16;
    int e = eid[b];
    v8f accS[4] = {}, accE[4] = {};
    const __bf16* xbB = xb + (size_t)b * C_N * HW_N;

    // ---- 3x3 pass (K = 9 taps * 128 ch, 4 channel-blocks of 32) ----
    for (int cblk = 0; cblk < 4; ++cblk) {
        for (int q = tid; q < 9 * 2048; q += 256) {
            int n = q & 63, c = (q >> 6) & 31, t = q >> 11;
            int p = p0 + n;
            int y = p / W_N, xx = p - y * W_N;
            int yy = y + t / 3 - 1, xc = xx + t % 3 - 1;
            __bf16 v = (__bf16)0.f;
            if ((unsigned)yy < (unsigned)W_N && (unsigned)xc < (unsigned)W_N)
                v = xbB[(size_t)(cblk * 32 + c) * HW_N + yy * W_N + xc];
            tb[t][n][c] = v;
        }
        if (cblk < 3)
            __builtin_prefetch(xbB + (size_t)((cblk + 1) * 32) * HW_N + p0, 0, 0);
        __syncthreads();
        for (int t = 0; t < 9; ++t) {
            int kc = t * 4 + cblk;
            v16bf aS = *(const v16bf*)(packS + ((size_t)(wave * 36 + kc) * 32 + lane) * 16);
            #pragma unroll
            for (int sub = 0; sub < 4; ++sub) {
                v16bf bf = *(const v16bf*)&tb[t][sub * 16 + nOff][cOff];
                accS[sub] = __builtin_amdgcn_wmma_f32_16x16x32_bf16(
                    false, aS, false, bf, (short)0, accS[sub], false, false);
            }
            if (e == 0) {
                v16bf aC = *(const v16bf*)(packC + ((size_t)(wave * 36 + kc) * 32 + lane) * 16);
                #pragma unroll
                for (int sub = 0; sub < 4; ++sub) {
                    v16bf bf = *(const v16bf*)&tb[t][sub * 16 + nOff][cOff];
                    accE[sub] = __builtin_amdgcn_wmma_f32_16x16x32_bf16(
                        false, aC, false, bf, (short)0, accE[sub], false, false);
                }
            }
        }
        __syncthreads();
    }

    // ---- bayar 5x5 pass (expert 1 only; taps staged in groups of 9) ----
    if (e == 1) {
        for (int cblk = 0; cblk < 4; ++cblk) {
            for (int g = 0; g < 3; ++g) {
                int t0 = g * 9;
                int t1 = (t0 + 9 > 25) ? 25 : t0 + 9;
                int cnt = t1 - t0;
                for (int q = tid; q < cnt * 2048; q += 256) {
                    int n = q & 63, c = (q >> 6) & 31, t = t0 + (q >> 11);
                    int p = p0 + n;
                    int y = p / W_N, xx = p - y * W_N;
                    int yy = y + t / 5 - 2, xc = xx + t % 5 - 2;
                    __bf16 v = (__bf16)0.f;
                    if ((unsigned)yy < (unsigned)W_N && (unsigned)xc < (unsigned)W_N)
                        v = xbB[(size_t)(cblk * 32 + c) * HW_N + yy * W_N + xc];
                    tb[t - t0][n][c] = v;
                }
                __syncthreads();
                for (int t = t0; t < t1; ++t) {
                    int kc = t * 4 + cblk;
                    v16bf aB = *(const v16bf*)(packB + ((size_t)(wave * 100 + kc) * 32 + lane) * 16);
                    #pragma unroll
                    for (int sub = 0; sub < 4; ++sub) {
                        v16bf bf = *(const v16bf*)&tb[t - t0][sub * 16 + nOff][cOff];
                        accE[sub] = __builtin_amdgcn_wmma_f32_16x16x32_bf16(
                            false, aB, false, bf, (short)0, accE[sub], false, false);
                    }
                }
                __syncthreads();
            }
        }
    }

    // C/D layout: VGPR r -> M = r + 8*(lane>=16), N = lane&15
    #pragma unroll
    for (int sub = 0; sub < 4; ++sub) {
        #pragma unroll
        for (int r = 0; r < 8; ++r) {
            int o = wave * 16 + r + ((lane & 16) ? 8 : 0);
            float v = accS[sub][r];
            if (e <= 1) v += accE[sub][r];
            out[(size_t)(b * C_N + o) * HW_N + p0 + sub * 16 + nOff] = v;
        }
    }
}

// ---------------------------------------------------------------------------
// Depthwise SRM (3 filters, clip +-3) and HF laplacian.
// Mids stored channel-contiguous: srm (B,HW,384), hf (B,HW,128) -> TDM tiles.
// Thread map is channel-fastest so mid writes coalesce.
__global__ void k_dw(const float* __restrict__ x, const float* __restrict__ srmK,
                     const float* __restrict__ hfK, __bf16* __restrict__ srm_mid,
                     __bf16* __restrict__ hf_mid) {
    for (int idx = blockIdx.x * blockDim.x + threadIdx.x; idx < NPIX;
         idx += gridDim.x * blockDim.x) {
        int c = idx & 127;
        int rest = idx >> 7;
        int p = rest % HW_N;
        int b = rest / HW_N;
        int y = p / W_N, xx = p - y * W_N;
        const float* xp = x + (size_t)(b * C_N + c) * HW_N;
        float win[25];
        #pragma unroll
        for (int t = 0; t < 25; ++t) {
            int yy = y + t / 5 - 2, xc = xx + t % 5 - 2;
            win[t] = ((unsigned)yy < (unsigned)W_N && (unsigned)xc < (unsigned)W_N)
                         ? xp[yy * W_N + xc] : 0.f;
        }
        const float* k0 = srmK + (size_t)(3 * c + 0) * 25;
        const float* k1 = srmK + (size_t)(3 * c + 1) * 25;
        const float* k2 = srmK + (size_t)(3 * c + 2) * 25;
        float s0 = 0.f, s1 = 0.f, s2 = 0.f;
        #pragma unroll
        for (int t = 0; t < 25; ++t) {
            s0 += win[t] * k0[t]; s1 += win[t] * k1[t]; s2 += win[t] * k2[t];
        }
        s0 = fminf(3.f, fmaxf(-3.f, s0));
        s1 = fminf(3.f, fmaxf(-3.f, s1));
        s2 = fminf(3.f, fmaxf(-3.f, s2));
        size_t sb = ((size_t)b * HW_N + p) * 384 + 3 * c;
        srm_mid[sb]     = (__bf16)s0;
        srm_mid[sb + 1] = (__bf16)s1;
        srm_mid[sb + 2] = (__bf16)s2;
        const float* kh = hfK + (size_t)c * 9;
        float h = 0.f;
        #pragma unroll
        for (int t = 0; t < 9; ++t) {
            int dy = t / 3 - 1, dx = t % 3 - 1;
            h += win[(dy + 2) * 5 + dx + 2] * kh[t];
        }
        hf_mid[((size_t)b * HW_N + p) * 128 + c] = (__bf16)h;
    }
}

// ---------------------------------------------------------------------------
// Dense 1x1 WMMA GEMMs for e2 (K=384) and e3 (K=128); B-tiles staged by the
// Tensor Data Mover, double-buffered on TENSORcnt; raw (pre-BN) bf16 outputs.
__global__ void __launch_bounds__(256, 2)
k_conv1x1(const __bf16* __restrict__ srm_mid, const __bf16* __restrict__ hf_mid,
          const __bf16* __restrict__ packR, const __bf16* __restrict__ packH,
          __bf16* __restrict__ e2raw, __bf16* __restrict__ e3raw) {
    __shared__ __align__(32) __bf16 tb[2][64][32];
    int wg = blockIdx.x;
    int b = wg / NT_N, nt = wg % NT_N;
    int p0 = nt * 64;
    int tid = threadIdx.x, lane = tid & 31, wave = tid >> 5;
    int nOff = lane & 15, cOff = lane & 16;
    unsigned lds[2] = { (unsigned)(uintptr_t)&tb[0][0][0],
                        (unsigned)(uintptr_t)&tb[1][0][0] };
    v8f acc2[4] = {}, acc3[4] = {};

    // ---- e2: K = 384, 12 chunks of 32 channels ----
    const __bf16* base2 = srm_mid + ((size_t)b * HW_N + p0) * 384;
    if (wave == 0) tdm_load_2d(lds[0], base2, 32, 64, 384);
    for (int kc = 0; kc < 12; ++kc) {
        if (wave == 0) __builtin_amdgcn_s_wait_tensorcnt(0);
        __syncthreads();
        if (wave == 0 && kc + 1 < 12)
            tdm_load_2d(lds[(kc + 1) & 1], base2 + (kc + 1) * 32, 32, 64, 384);
        const __bf16* tbuf = &tb[kc & 1][0][0];
        v16bf aR = *(const v16bf*)(packR + ((size_t)(wave * 12 + kc) * 32 + lane) * 16);
        #pragma unroll
        for (int sub = 0; sub < 4; ++sub) {
            v16bf bf = *(const v16bf*)(tbuf + (sub * 16 + nOff) * 32 + cOff);
            acc2[sub] = __builtin_amdgcn_wmma_f32_16x16x32_bf16(
                false, aR, false, bf, (short)0, acc2[sub], false, false);
        }
        __syncthreads();
    }

    // ---- e3: K = 128, 4 chunks of 32 channels ----
    const __bf16* base3 = hf_mid + ((size_t)b * HW_N + p0) * 128;
    if (wave == 0) tdm_load_2d(lds[0], base3, 32, 64, 128);
    for (int kc = 0; kc < 4; ++kc) {
        if (wave == 0) __builtin_amdgcn_s_wait_tensorcnt(0);
        __syncthreads();
        if (wave == 0 && kc + 1 < 4)
            tdm_load_2d(lds[(kc + 1) & 1], base3 + (kc + 1) * 32, 32, 64, 128);
        const __bf16* tbuf = &tb[kc & 1][0][0];
        v16bf aH = *(const v16bf*)(packH + ((size_t)(wave * 4 + kc) * 32 + lane) * 16);
        #pragma unroll
        for (int sub = 0; sub < 4; ++sub) {
            v16bf bf = *(const v16bf*)(tbuf + (sub * 16 + nOff) * 32 + cOff);
            acc3[sub] = __builtin_amdgcn_wmma_f32_16x16x32_bf16(
                false, aH, false, bf, (short)0, acc3[sub], false, false);
        }
        __syncthreads();
    }

    #pragma unroll
    for (int sub = 0; sub < 4; ++sub) {
        #pragma unroll
        for (int r = 0; r < 8; ++r) {
            int o = wave * 16 + r + ((lane & 16) ? 8 : 0);
            size_t a = (size_t)(b * C_N + o) * HW_N + p0 + sub * 16 + nOff;
            e2raw[a] = (__bf16)acc2[sub][r];
            e3raw[a] = (__bf16)acc3[sub][r];
        }
    }
}

// ---------------------------------------------------------------------------
// BN statistics over (B,H,W) per channel for e2 / e3; store scale & shift.
__global__ void k_bnstat(const __bf16* __restrict__ e2raw, const __bf16* __restrict__ e3raw,
                         const float* __restrict__ g2, const float* __restrict__ b2,
                         const float* __restrict__ g3, const float* __restrict__ b3,
                         float* __restrict__ bn) {
    int ch = blockIdx.x & 127, ex = blockIdx.x >> 7;
    const __bf16* src = ex ? e3raw : e2raw;
    float s = 0.f, ss = 0.f;
    for (int i = threadIdx.x; i < B_N * HW_N; i += blockDim.x) {
        int b = i / HW_N, p = i - (i / HW_N) * HW_N;
        float v = (float)src[(size_t)(b * C_N + ch) * HW_N + p];
        s += v; ss += v * v;
    }
    __shared__ float rs[256], rq[256];
    rs[threadIdx.x] = s; rq[threadIdx.x] = ss;
    __syncthreads();
    for (int o = 128; o > 0; o >>= 1) {
        if ((int)threadIdx.x < o) { rs[threadIdx.x] += rs[threadIdx.x + o];
                                    rq[threadIdx.x] += rq[threadIdx.x + o]; }
        __syncthreads();
    }
    if (threadIdx.x == 0) {
        float n = (float)(B_N * HW_N);
        float m = rs[0] / n;
        float var = rq[0] / n - m * m;
        float gamma = ex ? g3[ch] : g2[ch];
        float beta  = ex ? b3[ch] : b2[ch];
        float sc = gamma * rsqrtf(var + 1e-5f);
        bn[ex * 256 + ch]       = sc;
        bn[ex * 256 + 128 + ch] = beta - m * sc;
    }
}

// ---------------------------------------------------------------------------
// Add normalized+ReLU e2/e3 contribution for batches that selected them.
__global__ void k_mix(const __bf16* __restrict__ e2raw, const __bf16* __restrict__ e3raw,
                      const float* __restrict__ bn, const int* __restrict__ eid,
                      float* __restrict__ out) {
    for (int idx = blockIdx.x * blockDim.x + threadIdx.x; idx < NPIX;
         idx += gridDim.x * blockDim.x) {
        int bc = idx / HW_N;
        int c = bc & 127, b = bc >> 7;
        int e = eid[b];
        if (e == 2) {
            float v = (float)e2raw[idx] * bn[c] + bn[128 + c];
            out[idx] += fmaxf(v, 0.f);
        } else if (e == 3) {
            float v = (float)e3raw[idx] * bn[256 + c] + bn[384 + c];
            out[idx] += fmaxf(v, 0.f);
        }
    }
}

// ---------------------------------------------------------------------------
extern "C" void kernel_launch(void* const* d_in, const int* in_sizes, int n_in,
                              void* d_out, int out_size, void* d_ws, size_t ws_size,
                              hipStream_t stream) {
    (void)in_sizes; (void)n_in; (void)out_size; (void)ws_size;
    const float* x        = (const float*)d_in[0];
    const float* Wg       = (const float*)d_in[1];
    const float* cd_w     = (const float*)d_in[2];
    const float* bayar_w  = (const float*)d_in[3];
    const float* srm_k    = (const float*)d_in[4];
    const float* srm_ow   = (const float*)d_in[5];
    const float* srm_g    = (const float*)d_in[6];
    const float* srm_b    = (const float*)d_in[7];
    const float* hf_k     = (const float*)d_in[8];
    const float* hf_ow    = (const float*)d_in[9];
    const float* hf_g     = (const float*)d_in[10];
    const float* hf_b     = (const float*)d_in[11];
    const float* shared_w = (const float*)d_in[12];

    char* ws = (char*)d_ws;
    __bf16* xbf     = (__bf16*)(ws + OFS_XBF);
    __bf16* packS   = (__bf16*)(ws + OFS_PACKS);
    __bf16* packC   = (__bf16*)(ws + OFS_PACKC);
    __bf16* packB   = (__bf16*)(ws + OFS_PACKB);
    __bf16* packR   = (__bf16*)(ws + OFS_PACKR);
    __bf16* packH   = (__bf16*)(ws + OFS_PACKH);
    __bf16* srm_mid = (__bf16*)(ws + OFS_SRM);
    __bf16* hf_mid  = (__bf16*)(ws + OFS_HFM);
    __bf16* e2raw   = (__bf16*)(ws + OFS_E2);
    __bf16* e3raw   = (__bf16*)(ws + OFS_E3);
    float*  bn      = (float*)(ws + OFS_BN);
    int*    eid     = (int*)(ws + OFS_EID);

    float* out  = (float*)d_out;
    float* tail = out + (size_t)NPIX;       // aux(1) ex(4) imp(4) weights(128)
    float* outw = tail + 9;

    k_convert_bf16<<<4096, 256, 0, stream>>>(x, xbf);
    k_pack<<<512, 256, 0, stream>>>(shared_w, packS, 0, 36);
    k_pack<<<512, 256, 0, stream>>>(cd_w,     packC, 1, 36);
    k_pack<<<1024, 256, 0, stream>>>(bayar_w, packB, 2, 100);
    k_pack<<<256, 256, 0, stream>>>(srm_ow,   packR, 3, 12);
    k_pack<<<128, 256, 0, stream>>>(hf_ow,    packH, 4, 4);

    k_gate<<<B_N, 256, 0, stream>>>(x, Wg, outw, eid);
    k_gate_fin<<<1, 32, 0, stream>>>(outw, tail);

    k_conv_main<<<B_N * NT_N, 256, 0, stream>>>(xbf, packS, packC, packB, eid, out);

    k_dw<<<4096, 256, 0, stream>>>(x, srm_k, hf_k, srm_mid, hf_mid);
    k_conv1x1<<<B_N * NT_N, 256, 0, stream>>>(srm_mid, hf_mid, packR, packH, e2raw, e3raw);
    k_bnstat<<<256, 256, 0, stream>>>(e2raw, e3raw, srm_g, srm_b, hf_g, hf_b, bn);
    k_mix<<<4096, 256, 0, stream>>>(e2raw, e3raw, bn, eid, out);
}
